// PointNetSetAbstraction_19413252178083
// MI455X (gfx1250) — matrix operand order repaired and checked by reading, not compiled, not run on hardware
//
#include <hip/hip_runtime.h>

// ---------------------------------------------------------------------------
// PointNet++ Set Abstraction for MI455X (gfx1250, wave32, WMMA)
//   K1: FPS             — 1 workgroup/batch, dist[] in LDS, shuffle+LDS argmax
//   K2: knn+group       — per-centroid top-32 insertion select (LDS), builds
//                         fp16 "combined" A-matrix (M x 64) for GEMM1
//   K3: GEMM1 (WMMA f16)— Y1 = A@W1 + b1, channel sum/sumsq via LDS+atomics
//   K4: BN finalize     — scale/bias from sums
//   K5: GEMM2 (WMMA f16)— BN1+ReLU fused into A-load, Y2 = A2@W2 + b2 + stats
//   K6: BN finalize (2)
//   K7: BN2+ReLU+max_K  — writes new_feats
// Precision: f16 matmul inputs, f32 accumulation & statistics.
// ---------------------------------------------------------------------------

typedef _Float16 half_t;
typedef __attribute__((ext_vector_type(16))) _Float16 v16h;
typedef __attribute__((ext_vector_type(8)))  float    v8f;

#define BB    8
#define NN    8192
#define SS    1024
#define KK    32
#define DD    3
#define CF    61
#define CIN   64            // 3 + 61
#define COUT  128
#define MTOT  (BB*SS*KK)    // 262144
#define RAD2  0.04f         // 0.2^2
#define EPSV  1e-5f

union V16 { v16h v; float4 q[2]; half_t h[16]; };
union V8F { v8f  v; float  f[8]; };
union H8  { float4 q; half_t h[8]; };

// ---------------------------------------------------------------- K1: FPS ---
__global__ __launch_bounds__(1024) void fps_kernel(
    const float* __restrict__ xyz, const int* __restrict__ finit,
    float* __restrict__ new_xyz)
{
    __shared__ float dist[NN];       // 32 KB
    __shared__ float sC[3];
    __shared__ float redD[32];
    __shared__ int   redI[32];
    __shared__ int   sFar;

    const int b = blockIdx.x;
    const int t = threadIdx.x;
    const float* bx = xyz + (size_t)b * NN * 3;

    for (int j = t; j < NN; j += 1024) dist[j] = 1e10f;
    if (t == 0) sFar = finit[b];
    __syncthreads();

    for (int step = 0; step < SS; ++step) {
        if (t == 0) {
            const int far = sFar;
            const float cx = bx[far*3+0], cy = bx[far*3+1], cz = bx[far*3+2];
            sC[0] = cx; sC[1] = cy; sC[2] = cz;
            float* o = new_xyz + ((size_t)b*SS + step)*3;
            o[0] = cx; o[1] = cy; o[2] = cz;         // emit centroid coords
        }
        __syncthreads();
        const float cx = sC[0], cy = sC[1], cz = sC[2];

        float bestD = -1.0f; int bestI = 0x7fffffff;
        for (int j = t; j < NN; j += 1024) {
            const float dx = bx[j*3+0]-cx, dy = bx[j*3+1]-cy, dz = bx[j*3+2]-cz;
            float d = dx*dx + dy*dy + dz*dz;
            d = fminf(dist[j], d);
            dist[j] = d;
            if (d > bestD || (d == bestD && j < bestI)) { bestD = d; bestI = j; }
        }
        // wave32 reduce (argmax, ties -> lowest index like jnp.argmax)
        #pragma unroll
        for (int off = 16; off > 0; off >>= 1) {
            const float od = __shfl_xor(bestD, off);
            const int   oi = __shfl_xor(bestI, off);
            if (od > bestD || (od == bestD && oi < bestI)) { bestD = od; bestI = oi; }
        }
        const int wid = t >> 5;
        if ((t & 31) == 0) { redD[wid] = bestD; redI[wid] = bestI; }
        __syncthreads();
        if (wid == 0) {
            bestD = redD[t & 31]; bestI = redI[t & 31];
            #pragma unroll
            for (int off = 16; off > 0; off >>= 1) {
                const float od = __shfl_xor(bestD, off);
                const int   oi = __shfl_xor(bestI, off);
                if (od > bestD || (od == bestD && oi < bestI)) { bestD = od; bestI = oi; }
            }
            if (t == 0) sFar = bestI;
        }
        __syncthreads();
    }
}

// -------------------------------------------------------- K2: knn + group ---
#define KNN_T 64
__global__ __launch_bounds__(KNN_T) void knn_group_kernel(
    const float* __restrict__ xyz, const float* __restrict__ feat,
    const float* __restrict__ new_xyz, half_t* __restrict__ combined)
{
    __shared__ float kD[KNN_T][KK+1];   // +1 stagger vs 64 banks
    __shared__ int   kI[KNN_T][KK+1];

    const int t   = threadIdx.x;
    const int cid = blockIdx.x * KNN_T + t;       // 0..8191
    const int b   = cid / SS;
    const float* bx = xyz  + (size_t)b * NN * 3;
    const float* bf = feat + (size_t)b * NN * CF;

    const float cx = new_xyz[(size_t)cid*3+0];
    const float cy = new_xyz[(size_t)cid*3+1];
    const float cz = new_xyz[(size_t)cid*3+2];
    const float cs = cx*cx + cy*cy + cz*cz;

    float* D = kD[t]; int* I = kI[t];
    #pragma unroll
    for (int k = 0; k < KK; ++k) { D[k] = 3.0e38f; I[k] = 0; }

    // stable top-32 by squared distance (order-equivalent to reference sqrt)
    for (int j = 0; j < NN; ++j) {
        const float px = bx[j*3+0], py = bx[j*3+1], pz = bx[j*3+2];
        float d2 = cs + (px*px + py*py + pz*pz) - 2.0f*(cx*px + cy*py + cz*pz);
        d2 = fmaxf(d2, 0.0f);
        if (d2 < D[KK-1]) {
            int p = KK - 1;
            while (p > 0 && D[p-1] > d2) { D[p] = D[p-1]; I[p] = I[p-1]; --p; }
            D[p] = d2; I[p] = j;
        }
    }
    // build combined rows: [dx,dy,dz, feat(61)] as f16 (out-of-radius -> idx 0)
    for (int k = 0; k < KK; ++k) {
        const int g = (D[k] > RAD2) ? 0 : I[k];
        half_t* row = combined + ((size_t)cid*KK + k) * CIN;
        row[0] = (half_t)(bx[g*3+0] - cx);
        row[1] = (half_t)(bx[g*3+1] - cy);
        row[2] = (half_t)(bx[g*3+2] - cz);
        const float* f = bf + (size_t)g * CF;
        #pragma unroll 1
        for (int c = 0; c < CF; ++c) row[3+c] = (half_t)f[c];
    }
}

// -------------------------------------------------------------- K3: GEMM1 ---
// Y1[M,128] = combined[M,64] @ W1[64,128] + b1 ; accumulate ch sum/sumsq
__global__ __launch_bounds__(256) void gemm1_kernel(
    const half_t* __restrict__ Acomb, const float* __restrict__ W1,
    const float* __restrict__ b1, half_t* __restrict__ Y1,
    float* __restrict__ gSum, float* __restrict__ gSq)
{
    constexpr int AST = CIN + 8;                  // 72 f16 row stride (144B, 16B-mult)
    __shared__ half_t As[128 * AST];              // 18.4 KB
    __shared__ half_t Bs[COUT * AST];             // W1^T, [n][k], 18.4 KB
    __shared__ float  ldsSum[COUT], ldsSq[COUT];

    const int tid = threadIdx.x;
    if (tid < COUT) { ldsSum[tid] = 0.f; ldsSq[tid] = 0.f; }
    const size_t m0 = (size_t)blockIdx.x * 128;

    for (int i = tid; i < 1024; i += 256) {       // 128 rows x 8 float4
        const int row = i >> 3, c8 = i & 7;
        *(float4*)&As[row*AST + c8*8] =
            *(const float4*)&Acomb[(m0+row)*CIN + c8*8];
    }
    for (int i = tid; i < CIN*COUT; i += 256) {   // transpose W1 into LDS as f16
        const int k = i >> 7, n = i & 127;
        Bs[n*AST + k] = (half_t)W1[i];
    }
    __syncthreads();

    const int wave = tid >> 5, lane = tid & 31;
    const int hf = lane >> 4, l16 = lane & 15;
    const int n0 = wave * 16;
    const int ch = n0 + l16;                      // lane's output channel
    const float bias = b1[ch];
    float sum = 0.f, sq = 0.f;

    #pragma unroll 1
    for (int msub = 0; msub < 8; ++msub) {
        const int mrow = msub*16 + l16;
        V8F c;
        #pragma unroll
        for (int r = 0; r < 8; ++r) c.f[r] = 0.f;
        #pragma unroll
        for (int ks = 0; ks < 2; ++ks) {
            const int k0 = ks * 32;
            V16 a, w;
            // A 16x32: lane<16 K=0..7,16..23 ; lane>=16 shifted +8
            a.q[0] = *(const float4*)&As[mrow*AST + k0 + 8*hf];
            a.q[1] = *(const float4*)&As[mrow*AST + k0 + 16 + 8*hf];
            // B 32x16: lane<16 K=0..7,16..23 ; lane>=16 shifted +8 (N-major tile)
            w.q[0] = *(const float4*)&Bs[ch*AST + k0 + 8*hf];
            w.q[1] = *(const float4*)&Bs[ch*AST + k0 + 16 + 8*hf];
            c.v = __builtin_amdgcn_wmma_f32_16x16x32_f16(
                      false, a.v, false, w.v, (short)0, c.v, false, false);
        }
        #pragma unroll
        for (int r = 0; r < 8; ++r) {
            const float y = c.f[r] + bias;
            sum += y; sq += y*y;
            const size_t gm = m0 + msub*16 + r + 8*hf;   // C/D: M = r + 8*half
            Y1[gm*COUT + ch] = (half_t)y;
        }
    }
    atomicAdd(&ldsSum[ch], sum);
    atomicAdd(&ldsSq[ch],  sq);
    __syncthreads();
    if (tid < COUT) { atomicAdd(&gSum[tid], ldsSum[tid]); atomicAdd(&gSq[tid], ldsSq[tid]); }
}

// ------------------------------------------------------- K4: BN finalize ----
__global__ __launch_bounds__(COUT) void bn_finalize_kernel(
    const float* __restrict__ sum, const float* __restrict__ sq,
    const float* __restrict__ g, const float* __restrict__ be,
    float* __restrict__ scale, float* __restrict__ bias)
{
    const int c = threadIdx.x;
    const float inv = 1.0f / (float)MTOT;
    const float mean = sum[c] * inv;
    const float var  = sq[c] * inv - mean*mean;   // biased, like jnp.var
    const float sc   = rsqrtf(var + EPSV) * g[c];
    scale[c] = sc;
    bias[c]  = be[c] - mean * sc;
}

// -------------------------------------------------------------- K5: GEMM2 ---
// A2 = relu(BN1(Y1)) fused in A-load ; Y2 = A2 @ W2 + b2 ; stats for BN2
__global__ __launch_bounds__(256) void gemm2_kernel(
    const half_t* __restrict__ Y1, const float* __restrict__ W2,
    const float* __restrict__ b2, const float* __restrict__ scale1,
    const float* __restrict__ bias1, half_t* __restrict__ Y2,
    float* __restrict__ gSum, float* __restrict__ gSq)
{
    constexpr int AST = COUT + 8;                 // 136 f16 stride (272B, 16B-mult)
    __shared__ half_t As[64 * AST];               // 17.4 KB
    __shared__ half_t Bs[COUT * AST];             // 34.8 KB
    __shared__ float  ldsSum[COUT], ldsSq[COUT];

    const int tid = threadIdx.x;
    if (tid < COUT) { ldsSum[tid] = 0.f; ldsSq[tid] = 0.f; }
    const size_t m0 = (size_t)blockIdx.x * 64;

    for (int i = tid; i < 1024; i += 256) {       // 64 rows x 16 float4, BN1+ReLU
        const int row = i >> 4, c8 = i & 15;
        H8 in, out;
        in.q = *(const float4*)&Y1[(m0+row)*COUT + c8*8];
        #pragma unroll
        for (int e = 0; e < 8; ++e) {
            const int ch = c8*8 + e;
            const float y = (float)in.h[e] * scale1[ch] + bias1[ch];
            out.h[e] = (half_t)fmaxf(y, 0.f);
        }
        *(float4*)&As[row*AST + c8*8] = out.q;
    }
    for (int i = tid; i < COUT*COUT; i += 256) {  // transpose W2 as f16
        const int k = i >> 7, n = i & 127;
        Bs[n*AST + k] = (half_t)W2[i];
    }
    __syncthreads();

    const int wave = tid >> 5, lane = tid & 31;
    const int hf = lane >> 4, l16 = lane & 15;
    const int n0 = wave * 16;
    const int ch = n0 + l16;
    const float bias = b2[ch];
    float sum = 0.f, sq = 0.f;

    #pragma unroll 1
    for (int msub = 0; msub < 4; ++msub) {
        const int mrow = msub*16 + l16;
        V8F c;
        #pragma unroll
        for (int r = 0; r < 8; ++r) c.f[r] = 0.f;
        #pragma unroll
        for (int ks = 0; ks < 4; ++ks) {
            const int k0 = ks * 32;
            V16 a, w;
            a.q[0] = *(const float4*)&As[mrow*AST + k0 + 8*hf];
            a.q[1] = *(const float4*)&As[mrow*AST + k0 + 16 + 8*hf];
            w.q[0] = *(const float4*)&Bs[ch*AST + k0 + 8*hf];
            w.q[1] = *(const float4*)&Bs[ch*AST + k0 + 16 + 8*hf];
            c.v = __builtin_amdgcn_wmma_f32_16x16x32_f16(
                      false, a.v, false, w.v, (short)0, c.v, false, false);
        }
        #pragma unroll
        for (int r = 0; r < 8; ++r) {
            const float y = c.f[r] + bias;
            sum += y; sq += y*y;
            const size_t gm = m0 + msub*16 + r + 8*hf;
            Y2[gm*COUT + ch] = (half_t)y;
        }
    }
    atomicAdd(&ldsSum[ch], sum);
    atomicAdd(&ldsSq[ch],  sq);
    __syncthreads();
    if (tid < COUT) { atomicAdd(&gSum[tid], ldsSum[tid]); atomicAdd(&gSq[tid], ldsSq[tid]); }
}

// -------------------------------------------- K7: BN2 + ReLU + max over K ---
__global__ __launch_bounds__(COUT) void maxpool_kernel(
    const half_t* __restrict__ Y2, const float* __restrict__ scale2,
    const float* __restrict__ bias2, float* __restrict__ out_feats)
{
    const int bs = blockIdx.x;                    // 0..8191
    const int c  = threadIdx.x;                   // channel
    const float sc = scale2[c], bi = bias2[c];
    const half_t* base = Y2 + (size_t)bs * KK * COUT + c;
    float m = 0.0f;                               // relu output >= 0
    #pragma unroll
    for (int k = 0; k < KK; ++k) {
        float y = (float)base[(size_t)k*COUT] * sc + bi;
        y = fmaxf(y, 0.f);
        m = fmaxf(m, y);
    }
    out_feats[(size_t)bs*COUT + c] = m;
}

// ------------------------------------------------------------------ launch --
extern "C" void kernel_launch(void* const* d_in, const int* in_sizes, int n_in,
                              void* d_out, int out_size, void* d_ws, size_t ws_size,
                              hipStream_t stream)
{
    (void)in_sizes; (void)n_in; (void)out_size; (void)ws_size;
    const float* xyz   = (const float*)d_in[0];
    const float* feat  = (const float*)d_in[1];
    const int*   finit = (const int*)  d_in[2];
    const float* W1    = (const float*)d_in[3];
    const float* b1    = (const float*)d_in[4];
    const float* g1    = (const float*)d_in[5];
    const float* be1   = (const float*)d_in[6];
    const float* W2    = (const float*)d_in[7];
    const float* b2    = (const float*)d_in[8];
    const float* g2    = (const float*)d_in[9];
    const float* be2   = (const float*)d_in[10];

    float* out_newxyz = (float*)d_out;                   // B*S*3
    float* out_feats  = (float*)d_out + (size_t)BB*SS*3; // B*S*128

    // workspace layout
    char* ws = (char*)d_ws;
    float* sum1 = (float*)(ws + 0);
    float* sq1  = (float*)(ws + 512);
    float* sum2 = (float*)(ws + 1024);
    float* sq2  = (float*)(ws + 1536);
    float* sc1  = (float*)(ws + 2048);
    float* bi1  = (float*)(ws + 2560);
    float* sc2  = (float*)(ws + 3072);
    float* bi2  = (float*)(ws + 3584);
    half_t* combined = (half_t*)(ws + 65536);                         // 32 MB
    half_t* Y1       = (half_t*)(ws + 65536 + (size_t)MTOT*CIN*2);    // 64 MB
    half_t* Y2       = (half_t*)((char*)Y1 + (size_t)MTOT*COUT*2);    // 64 MB

    hipMemsetAsync(ws, 0, 2048, stream);  // zero sum/sumsq accumulators

    fps_kernel<<<BB, 1024, 0, stream>>>(xyz, finit, out_newxyz);
    knn_group_kernel<<<(BB*SS)/KNN_T, KNN_T, 0, stream>>>(xyz, feat, out_newxyz, combined);
    gemm1_kernel<<<MTOT/128, 256, 0, stream>>>(combined, W1, b1, Y1, sum1, sq1);
    bn_finalize_kernel<<<1, COUT, 0, stream>>>(sum1, sq1, g1, be1, sc1, bi1);
    gemm2_kernel<<<MTOT/64, 256, 0, stream>>>(Y1, W2, b2, sc1, bi1, Y2, sum2, sq2);
    bn_finalize_kernel<<<1, COUT, 0, stream>>>(sum2, sq2, g2, be2, sc2, bi2);
    maxpool_kernel<<<BB*SS, COUT, 0, stream>>>(Y2, sc2, bi2, out_feats);
}
